// Vocoder_807453851730
// MI455X (gfx1250) — matrix operand order repaired and checked by reading, not compiled, not run on hardware
//
#include <hip/hip_runtime.h>
#include <hip/hip_bf16.h>

// ---------------- Problem constants (from reference) ----------------
#define PI_F 3.14159265358979f
constexpr int   kNFFT = 2048;
constexpr int   kHOP  = 512;
constexpr int   kPAD  = 1024;              // N_FFT/2
constexpr int   kB    = 4;
constexpr int   kT    = 400;               // frames per batch
constexpr int   kF    = 1025;              // onesided bins
constexpr int   kK2F  = 2050;              // 2*F (re/im interleaved)
constexpr int   kKP   = 2112;              // 2F padded to multiple of 64 (and 32)
constexpr int   kM    = kB * kT;           // 1600 GEMM rows (real)
constexpr int   kMP   = 1664;              // M padded to multiple of 128 (13 blocks)
constexpr int   kL    = kHOP * (kT - 1);   // 204288 output samples per batch
constexpr int   kLP   = kL + 2 * kPAD;     // 206336 padded length
constexpr int   kIter = 60;
constexpr float kMom  = 0.99f / 1.99f;     // torchaudio momentum normalization
constexpr float kEps  = 1e-16f;

constexpr int kLdsStride = 40;             // tile row stride in halves (80B = 20 banks,
                                           // 16 lanes * b128 -> banks 0,4,...,60: conflict-free)

typedef __attribute__((ext_vector_type(16))) _Float16 v16h;
typedef __attribute__((ext_vector_type(8)))  _Float16 v8h;
typedef __attribute__((ext_vector_type(8)))  float    v8f;

__device__ __forceinline__ float hann_w(int n) {
  return 0.5f - 0.5f * __cosf(2.0f * PI_F * (float)n / (float)kNFFT);
}

// ---------------- Weight-matrix generation ----------------
// WinvT: (2048 rows = output sample n) x (kKP cols = 2k | 2k+1); window folded in,
// 1/N folded into overlap-add normalize so f16 weights stay O(1).
__global__ void k_init_winvT(_Float16* __restrict__ W) {
  int idx = blockIdx.x * blockDim.x + threadIdx.x;
  if (idx >= kNFFT * kKP) return;
  int n = idx / kKP, kk = idx - n * kKP;
  float v = 0.0f;
  if (kk < kK2F) {
    int k = kk >> 1;
    float c = (k == 0 || k == kNFFT / 2) ? 1.0f : 2.0f;
    int r = (k * n) & (kNFFT - 1);                     // phase mod N (N = 2^11)
    float ph = 2.0f * PI_F * (float)r / (float)kNFFT;
    float s, co;
    __sincosf(ph, &s, &co);
    v = ((kk & 1) ? -s : co) * c * hann_w(n);
  }
  W[idx] = (_Float16)v;
}

// WfwdT: (kKP rows = 2k | 2k+1) x (2048 cols = sample j)
__global__ void k_init_wfwdT(_Float16* __restrict__ W) {
  int idx = blockIdx.x * blockDim.x + threadIdx.x;
  if (idx >= kKP * kNFFT) return;
  int kk = idx / kNFFT, j = idx & (kNFFT - 1);
  float v = 0.0f;
  if (kk < kK2F) {
    int k = kk >> 1;
    int r = (k * j) & (kNFFT - 1);
    float ph = 2.0f * PI_F * (float)r / (float)kNFFT;
    float s, co;
    __sincosf(ph, &s, &co);
    v = (kk & 1) ? -s : co;
  }
  W[idx] = (_Float16)v;
}

// Window-square overlap sum (<=4 overlapping frames per sample).
__global__ void k_init_wsq(float* __restrict__ wsq) {
  int j = blockIdx.x * blockDim.x + threadIdx.x;
  if (j >= kLP) return;
  float s = 0.0f;
  int tm = j >> 9;
  for (int dt = 0; dt < 4; ++dt) {
    int t = tm - dt;
    if (t < 0 || t >= kT) continue;
    int n = j - (t << 9);
    if (n < kNFFT) { float w = hann_w(n); s += w * w; }
  }
  wsq[j] = s;
}

__global__ void k_init_state(float* __restrict__ ang, float* __restrict__ tprev) {
  int idx = blockIdx.x * blockDim.x + threadIdx.x;
  if (idx >= kM * kF) return;
  int m = idx / kF, k = idx - m * kF;
  int base = m * kK2F + 2 * k;
  ang[base] = 1.0f; ang[base + 1] = 0.0f;     // angles0 = 1 + 0i
  tprev[base] = 0.0f; tprev[base + 1] = 0.0f;
}

__global__ void k_zero_half(_Float16* __restrict__ p, int n) {
  int idx = blockIdx.x * blockDim.x + threadIdx.x;
  if (idx < n) p[idx] = (_Float16)0.0f;
}

// S = mag * angles, interleaved re/im, f16, row stride kKP (pad cols/rows stay 0).
__global__ void k_build_S(const float* __restrict__ mag, const float* __restrict__ ang,
                          _Float16* __restrict__ S) {
  int idx = blockIdx.x * blockDim.x + threadIdx.x;
  if (idx >= kM * kF) return;
  int m = idx / kF, k = idx - m * kF;
  float s  = mag[idx];                        // input (B,T,F) flat == (m,k)
  float re = ang[m * kK2F + 2 * k];
  float im = ang[m * kK2F + 2 * k + 1];
  S[m * kKP + 2 * k]     = (_Float16)(s * re);
  S[m * kKP + 2 * k + 1] = (_Float16)(s * im);
}

// ---------------- WMMA GEMM with async-LDS double buffering ----------------
// C[Mp x N](f32) = A[Mp x K](f16) * Bt[N x K]^T(f16).  Mp multiple of 128, N of 64,
// K of 32.  Block = 8 waves; block tile 128x64; K-step 32.
// Tiles staged in LDS via GLOBAL_LOAD_ASYNC_TO_LDS_B128 (ASYNCcnt), drained with
// s_wait_asynccnt; B tile shared by all 8 waves (8x global-traffic reduction).
__global__ __launch_bounds__(256)
void k_wmma_gemm_lds(const _Float16* __restrict__ A, const _Float16* __restrict__ Bt,
                     float* __restrict__ C, int N, int K) {
  __shared__ _Float16 sA[2][128 * kLdsStride];
  __shared__ _Float16 sB[2][64 * kLdsStride];

  const int tid  = threadIdx.x;
  const int lane = tid & 31;
  const int wave = tid >> 5;
  const int hsel = lane >> 4;                  // 0: lanes 0-15, 1: lanes 16-31
  const int l15  = lane & 15;
  const int mBase = blockIdx.y * 128;
  const int n0    = blockIdx.x * 64;

  // Per-thread async-copy assignment: A rows 0..127 (2 x 16B each), B rows 0..63 (16B).
  const int arow = tid >> 1, acol = (tid & 1) * 16;      // halves
  const int brow = tid >> 2, bcol = (tid & 3) * 8;       // halves
  const _Float16* gA = A  + (size_t)(mBase + arow) * K + acol;
  const _Float16* gB = Bt + (size_t)(n0 + brow) * K + bcol;
  // Generic->LDS offset: low 32 bits of the flat shared address are the LDS byte offset.
  const unsigned laA = (unsigned)(size_t)&sA[0][arow * kLdsStride + acol];
  const unsigned laB = (unsigned)(size_t)&sB[0][brow * kLdsStride + bcol];
  const unsigned bufABytes = (unsigned)(128 * kLdsStride * 2);
  const unsigned bufBBytes = (unsigned)(64 * kLdsStride * 2);

  auto issue = [&](int buf, int k0) {
    unsigned a0 = laA + (unsigned)buf * bufABytes;
    unsigned b0 = laB + (unsigned)buf * bufBBytes;
    asm volatile("global_load_async_to_lds_b128 %0, %1, off"
                 :: "v"(a0), "v"(gA + k0) : "memory");
    asm volatile("global_load_async_to_lds_b128 %0, %1, off"
                 :: "v"(a0 + 16u), "v"(gA + k0 + 8) : "memory");
    asm volatile("global_load_async_to_lds_b128 %0, %1, off"
                 :: "v"(b0), "v"(gB + k0) : "memory");
  };

  v8f c0 = {}, c1 = {}, c2 = {}, c3 = {};
  const int NK = K / 32;
  issue(0, 0);
  for (int kt = 0; kt < NK; ++kt) {
    const int cur = kt & 1;
    asm volatile("s_wait_asynccnt 0x0" ::: "memory");
    __syncthreads();                     // publish copies; fence prev reads of 1-cur
    if (kt + 1 < NK) issue(1 - cur, (kt + 1) * 32);

    // A fragment (contiguous K, matching B): lanes 0-15 K=0..15, 16-31 K=16..31
    const _Float16* a0 = &sA[cur][(16 * wave + l15) * kLdsStride];
    v16h a;
    ((v8h*)&a)[0] = *(const v8h*)(a0 + hsel * 16);
    ((v8h*)&a)[1] = *(const v8h*)(a0 + hsel * 16 + 8);
    // B fragments: lane = output column, lanes 0-15 K-low half, 16-31 K-high half
    const _Float16* b0 = &sB[cur][l15 * kLdsStride + hsel * 16];
    v16h bv0 = *(const v16h*)(b0);
    v16h bv1 = *(const v16h*)(b0 + 16 * kLdsStride);
    v16h bv2 = *(const v16h*)(b0 + 32 * kLdsStride);
    v16h bv3 = *(const v16h*)(b0 + 48 * kLdsStride);

    c0 = __builtin_amdgcn_wmma_f32_16x16x32_f16(false, a, false, bv0, (short)0, c0, false, false);
    c1 = __builtin_amdgcn_wmma_f32_16x16x32_f16(false, a, false, bv1, (short)0, c1, false, false);
    c2 = __builtin_amdgcn_wmma_f32_16x16x32_f16(false, a, false, bv2, (short)0, c2, false, false);
    c3 = __builtin_amdgcn_wmma_f32_16x16x32_f16(false, a, false, bv3, (short)0, c3, false, false);
  }

  // 32-bit C/D layout: VGPR r -> row m0 + hsel*8 + r, col n0 + l15 (+tile*16)
  float* __restrict__ Cp = C + (size_t)(mBase + 16 * wave + hsel * 8) * N + n0 + l15;
#pragma unroll
  for (int r = 0; r < 8; ++r) {
    Cp[(size_t)r * N +  0] = c0[r];
    Cp[(size_t)r * N + 16] = c1[r];
    Cp[(size_t)r * N + 32] = c2[r];
    Cp[(size_t)r * N + 48] = c3[r];
  }
}

// Overlap-add (gather form, no atomics) + 1/N + wsq normalize -> core signal.
__global__ void k_ola_norm(const float* __restrict__ frames, const float* __restrict__ wsq,
                           float* __restrict__ out) {
  int idx = blockIdx.x * blockDim.x + threadIdx.x;
  if (idx >= kB * kL) return;
  int b = idx / kL, i = idx - b * kL;
  int j = i + kPAD;
  float s = 0.0f;
  int tm = j >> 9;
  for (int dt = 0; dt < 4; ++dt) {
    int t = tm - dt;
    if (t < 0 || t >= kT) continue;
    int n = j - (t << 9);
    if (n < kNFFT) s += frames[((size_t)(b * kT + t)) * kNFFT + n];
  }
  float w = wsq[j];
  float d = (w > 1e-11f) ? w : 1.0f;
  out[idx] = s * (1.0f / (float)kNFFT) / d;
}

// Reflect-pad + frame + window -> f16 GEMM A matrix for the STFT.
__global__ void k_build_frames(const float* __restrict__ ycore, _Float16* __restrict__ A) {
  int idx = blockIdx.x * blockDim.x + threadIdx.x;
  if (idx >= kM * kNFFT) return;
  int m = idx >> 11, n = idx & (kNFFT - 1);
  int b = m / kT, t = m - b * kT;
  int pos = t * kHOP + n - kPAD;
  if (pos < 0) pos = -pos;
  else if (pos >= kL) pos = 2 * kL - 2 - pos;
  A[idx] = (_Float16)(ycore[(size_t)b * kL + pos] * hann_w(n));
}

// a = rebuilt - mom*tprev; angles = a/(|a|+eps); tprev = rebuilt
__global__ void k_update_angles(const float* __restrict__ rb, float* __restrict__ ang,
                                float* __restrict__ tprev) {
  int idx = blockIdx.x * blockDim.x + threadIdx.x;
  if (idx >= kM * kF) return;
  int m = idx / kF, k = idx - m * kF;
  float re = rb[(size_t)m * kKP + 2 * k];
  float im = rb[(size_t)m * kKP + 2 * k + 1];
  int base = m * kK2F + 2 * k;
  float ar = re - kMom * tprev[base];
  float ai = im - kMom * tprev[base + 1];
  float mag = sqrtf(ar * ar + ai * ai) + kEps;
  ang[base]     = ar / mag;
  ang[base + 1] = ai / mag;
  tprev[base]     = re;
  tprev[base + 1] = im;
}

// ---------------- Host-side orchestration ----------------
static inline void* wsAlloc(unsigned char*& p, size_t bytes) {
  void* r = (void*)p;
  p += (bytes + 255) & ~(size_t)255;
  return r;
}

extern "C" void kernel_launch(void* const* d_in, const int* in_sizes, int n_in,
                              void* d_out, int out_size, void* d_ws, size_t ws_size,
                              hipStream_t stream) {
  const float* mag = (const float*)d_in[0];   // (B,T,F) float32
  float* outp = (float*)d_out;                // (B, kL) float32

  unsigned char* p = (unsigned char*)d_ws;
  _Float16* WinvT  = (_Float16*)wsAlloc(p, (size_t)kNFFT * kKP * 2);
  _Float16* WfwdT  = (_Float16*)wsAlloc(p, (size_t)kKP * kNFFT * 2);
  _Float16* Shalf  = (_Float16*)wsAlloc(p, (size_t)kMP * kKP * 2);
  _Float16* Afrm   = (_Float16*)wsAlloc(p, (size_t)kMP * kNFFT * 2);
  float*    frames = (float*)wsAlloc(p, (size_t)kMP * kNFFT * 4);
  float*    rebuilt= (float*)wsAlloc(p, (size_t)kMP * kKP * 4);
  float*    ang    = (float*)wsAlloc(p, (size_t)kM * kK2F * 4);
  float*    tprev  = (float*)wsAlloc(p, (size_t)kM * kK2F * 4);
  float*    ycore  = (float*)wsAlloc(p, (size_t)kB * kL * 4);
  float*    wsq    = (float*)wsAlloc(p, (size_t)kLP * 4);
  if ((size_t)(p - (unsigned char*)d_ws) > ws_size) return;  // scratch too small

  const int TB = 256;
  auto g = [](int n) { return dim3((n + 255) / 256); };

  // Setup — deterministic, re-run every launch (also re-zeroes pad rows/cols).
  k_init_winvT<<<g(kNFFT * kKP), TB, 0, stream>>>(WinvT);
  k_init_wfwdT<<<g(kKP * kNFFT), TB, 0, stream>>>(WfwdT);
  k_init_wsq  <<<g(kLP), TB, 0, stream>>>(wsq);
  k_init_state<<<g(kM * kF), TB, 0, stream>>>(ang, tprev);
  k_zero_half <<<g(kMP * kKP),  TB, 0, stream>>>(Shalf, kMP * kKP);
  k_zero_half <<<g(kMP * kNFFT),TB, 0, stream>>>(Afrm,  kMP * kNFFT);

  const dim3 gemmBlk(256);
  const dim3 gIstft(kNFFT / 64, kMP / 128);   // N=2048
  const dim3 gStft (kKP  / 64, kMP / 128);    // N=2112

  for (int it = 0; it < kIter; ++it) {
    k_build_S<<<g(kM * kF), TB, 0, stream>>>(mag, ang, Shalf);
    k_wmma_gemm_lds<<<gIstft, gemmBlk, 0, stream>>>(Shalf, WinvT, frames, kNFFT, kKP);
    k_ola_norm<<<g(kB * kL), TB, 0, stream>>>(frames, wsq, ycore);
    k_build_frames<<<g(kM * kNFFT), TB, 0, stream>>>(ycore, Afrm);
    k_wmma_gemm_lds<<<gStft, gemmBlk, 0, stream>>>(Afrm, WfwdT, rebuilt, kKP, kNFFT);
    k_update_angles<<<g(kM * kF), TB, 0, stream>>>(rebuilt, ang, tprev);
  }

  // Final ISTFT with converged phases, written straight to d_out.
  k_build_S<<<g(kM * kF), TB, 0, stream>>>(mag, ang, Shalf);
  k_wmma_gemm_lds<<<gIstft, gemmBlk, 0, stream>>>(Shalf, WinvT, frames, kNFFT, kKP);
  k_ola_norm<<<g(kB * kL), TB, 0, stream>>>(frames, wsq, outp);
}
